// CausalSelfAttention_64304250356401
// MI455X (gfx1250) — compile-verified, matrix operand order checked by reading, not run on hardware
//
#include <hip/hip_runtime.h>
#include <hip/hip_bf16.h>
#include <cstdint>
#include <cstddef>

// ---------------- problem constants (from reference) ----------------
#define BATCH 4
#define SEQ   1024
#define DIM   2048
#define NH    16
#define HD    128               // head dim (HD == HDE == 128)
#define MROWS (BATCH * SEQ)     // 4096 total rows

typedef __bf16 bf16;
typedef unsigned int u32;
typedef __attribute__((ext_vector_type(16))) __bf16 v16bf;
typedef __attribute__((ext_vector_type(8)))  __bf16 v8bf;
typedef __attribute__((ext_vector_type(4)))  __bf16 v4bf;
typedef __attribute__((ext_vector_type(8)))  float  v8f;
typedef __attribute__((ext_vector_type(4)))  u32    v4u;
typedef __attribute__((ext_vector_type(8)))  int    v8i;
typedef __attribute__((ext_vector_type(4)))  int    v4i;

static __device__ __forceinline__ v8f wmma_bf16(v16bf a, v16bf b, v8f c) {
  // D = A(16x32 bf16) * B(32x16 bf16) + C(16x16 f32)
  return __builtin_amdgcn_wmma_f32_16x16x32_bf16(false, a, false, b, (short)0, c,
                                                 false, false);
}

// Per-lane fragment gather for the 16-bit WMMA A/B layout:
// lane L covers row/col (L%16); K-runs of 8 at kbase=(L/16)*8 and kbase+16.
// p must point at (row, kbase) with 16-byte alignment.
static __device__ __forceinline__ v16bf load_frag16(const bf16* p) {
  v8bf lo = *(const v8bf*)(p);
  v8bf hi = *(const v8bf*)(p + 16);
  return __builtin_shufflevector(lo, hi, 0, 1, 2, 3, 4, 5, 6, 7,
                                 8, 9, 10, 11, 12, 13, 14, 15);
}

// ---------------------------------------------------------------------
// Tensor Data Mover: 2-D tile (rows x row_elems bf16) global -> LDS with
// LDS row padding (pad_amount DWORDs inserted every pad_interval DWORDs).
// Descriptor packing per CDNA5 ISA §8.3/8.4 (D# groups 0/1; rest = 0).
// This toolchain exposes the 6-arg builtin:
//   (uint32x4 g0, int32x8 g1, int32x4, int32x4, int32x8, i32 cpol)
// ---------------------------------------------------------------------
static __device__ __forceinline__ void tdm_load_2d_bf16(
    u32 lds_byte, const void* gptr, u32 rows, u32 row_elems,
    u32 row_stride_elems, u32 pad_interval_code, u32 pad_amount_code)
{
  const unsigned long long ga = (unsigned long long)(uintptr_t)gptr;
  v4u g0;
  g0[0] = 1u;                                        // count=1, no gather
  g0[1] = lds_byte;                                  // lds_addr
  g0[2] = (u32)ga;                                   // global_addr[31:0]
  g0[3] = (u32)((ga >> 32) & 0x1FFFFFFu) | (2u << 30); // addr[56:32] | type=2
  const u32 w0 = (1u << 16)                          // data_size = 2 bytes
               | (1u << 20)                          // pad_enable
               | (pad_interval_code << 22)           // pad interval (DWORDs)
               | (pad_amount_code << 25);            // pad amount  (DWORDs)
  v8i g1;
  g1[0] = (int)w0;
  g1[1] = (int)(row_elems << 16);        // tensor_dim0[15:0] @ bits 63:48
  g1[2] = (int)(rows << 16);             // tensor_dim1[15:0] @ bits 95:80
  g1[3] = (int)(row_elems << 16);        // tile_dim0 @ bits 127:112
  g1[4] = (int)rows;                     // tile_dim1 @ bits 143:128
  g1[5] = (int)row_stride_elems;         // tensor_dim0_stride[31:0]
  g1[6] = 0;
  g1[7] = 0;
  const v4i z4 = {0, 0, 0, 0};
  const v8i z8 = {0, 0, 0, 0, 0, 0, 0, 0};
  __builtin_amdgcn_tensor_load_to_lds(g0, g1, z4, z4, z8, 0);
}

// ======================================================================
// f32-in / bf16-WMMA / f32-accum GEMM:  C[M,N] = X[M,K] @ W[K,N]
// Double-buffered LDS software pipeline.
// MODE 0: store f32 to out[M,N]
// MODE 1: qk GEMM (N = 2*DIM): fused RoPE, split q/k, write bf16 (B,H,T,HD)
// MODE 2: v  GEMM (N = DIM)  : head transpose, write bf16 (B,H,T,HD)
// ======================================================================
#define BM 128
#define BN 128
#define BKK 32
#define LDK 40   // padded k-stride in halves (80 B: keeps 16 B alignment)

template <int MODE>
__global__ __launch_bounds__(256)
void gemm_bf16_kernel(const float* __restrict__ X, const float* __restrict__ W,
                      int N, int K,
                      float* __restrict__ out,
                      bf16* __restrict__ qws, bf16* __restrict__ kws,
                      bf16* __restrict__ vws,
                      const float* __restrict__ fcos,
                      const float* __restrict__ fsin)
{
  __shared__ bf16 lA[2][BM * LDK];   // [m][k]
  __shared__ bf16 lB[2][BN * LDK];   // [n][k]  (B tile stored transposed)

  const int tid  = threadIdx.x;
  const int lane = tid & 31;
  const int wv   = tid >> 5;         // wave 0..7
  const int wm   = wv >> 1;          // 0..3 : 32-row slab
  const int wn   = wv & 1;           // 0..1 : 64-col slab
  const int lr   = lane & 15;
  const int g8   = (lane >> 4) * 8;  // 0 or 8

  const int row0 = blockIdx.y * BM;
  const int col0 = blockIdx.x * BN;

  // ---- per-thread staging geometry (4 tasks each for A and B) ----
  const float* aSrc[4]; int aOff[4];
  const float* bSrc[4]; int bOff[4];
#pragma unroll
  for (int r = 0; r < 4; ++r) {
    const int t1 = tid + r * 256;            // 0..1023
    const int am = t1 >> 3, ak = t1 & 7;     // A: 128 rows x 8 k-quads
    aSrc[r] = X + (size_t)(row0 + am) * K + ak * 4;
    aOff[r] = am * LDK + ak * 4;
    const int bn = t1 & 127, bk = t1 >> 7;   // B: 8 k-quads x 128 cols
    bSrc[r] = W + (size_t)(bk * 4) * N + col0 + bn;
    bOff[r] = bn * LDK + bk * 4;
  }

  float4 ra[4];
  float  rb[4][4];

  auto loadg = [&](int kt) {
#pragma unroll
    for (int r = 0; r < 4; ++r) {
      ra[r] = *(const float4*)(aSrc[r] + (size_t)kt * BKK);
      __builtin_prefetch(aSrc[r] + (size_t)(kt + 1) * BKK, 0, 1);
    }
#pragma unroll
    for (int r = 0; r < 4; ++r) {
#pragma unroll
      for (int j = 0; j < 4; ++j)
        rb[r][j] = bSrc[r][((size_t)kt * BKK + j) * N];
      __builtin_prefetch(bSrc[r] + (size_t)(kt + 1) * BKK * N, 0, 1);
    }
  };
  auto storel = [&](int buf) {
#pragma unroll
    for (int r = 0; r < 4; ++r) {
      v4bf h;
      h[0] = (bf16)ra[r].x; h[1] = (bf16)ra[r].y;
      h[2] = (bf16)ra[r].z; h[3] = (bf16)ra[r].w;
      *(v4bf*)&lA[buf][aOff[r]] = h;
    }
#pragma unroll
    for (int r = 0; r < 4; ++r) {
      v4bf h;
      h[0] = (bf16)rb[r][0]; h[1] = (bf16)rb[r][1];
      h[2] = (bf16)rb[r][2]; h[3] = (bf16)rb[r][3];
      *(v4bf*)&lB[buf][bOff[r]] = h;
    }
  };

  const v8f vzero = {0.f, 0.f, 0.f, 0.f, 0.f, 0.f, 0.f, 0.f};
  v8f acc[2][4];
  for (int i = 0; i < 2; ++i)
    for (int j = 0; j < 4; ++j) acc[i][j] = vzero;

  const int nk = K / BKK;
  loadg(0);
  storel(0);
  __syncthreads();

  for (int kt = 0; kt < nk; ++kt) {
    const int cur = kt & 1;
    if (kt + 1 < nk) loadg(kt + 1);       // global loads overlap WMMA below

    const v16bf a0 = load_frag16(&lA[cur][(wm * 32 + 0  + lr) * LDK + g8]);
    const v16bf a1 = load_frag16(&lA[cur][(wm * 32 + 16 + lr) * LDK + g8]);
#pragma unroll
    for (int nt = 0; nt < 4; ++nt) {
      const v16bf b = load_frag16(&lB[cur][(wn * 64 + nt * 16 + lr) * LDK + g8]);
      acc[0][nt] = wmma_bf16(a0, b, acc[0][nt]);
      acc[1][nt] = wmma_bf16(a1, b, acc[1][nt]);
    }

    if (kt + 1 < nk) storel(cur ^ 1);     // waits loadcnt here, after WMMAs
    __syncthreads();
  }

  // ------------------------------ epilogue ------------------------------
  for (int mt = 0; mt < 2; ++mt) {
    const int mbase = row0 + wm * 32 + mt * 16 + g8;  // + j for C row
    for (int nt = 0; nt < 4; ++nt) {
      v8f c = acc[mt][nt];
      const int n = col0 + wn * 64 + nt * 16 + lr;
      if constexpr (MODE == 0) {
#pragma unroll
        for (int j = 0; j < 8; ++j)
          out[(size_t)(mbase + j) * N + n] = c[j];
      } else if constexpr (MODE == 1) {
        // fused RoPE; even/odd n are the re/im pair, living in adjacent lanes
        const int isk   = n >> 11;          // 0 = q half, 1 = k half
        const int dfull = n & (DIM - 1);
        const int hh    = dfull >> 7;
        const int hd    = dfull & (HD - 1);
        const int fi    = hd >> 1;
        bf16* dst = isk ? kws : qws;
#pragma unroll
        for (int j = 0; j < 8; ++j) {
          const int m = mbase + j;
          const int b = m >> 10;
          const int t = m & (SEQ - 1);
          const float cc  = fcos[t * (HD / 2) + fi];
          const float ss  = fsin[t * (HD / 2) + fi];
          const float val = c[j];
          const float oth = __shfl_xor(val, 1, 32);
          const float o = ((lane & 1) == 0) ? (val * cc - oth * ss)
                                            : (oth * ss + val * cc);
          dst[(((size_t)(b * NH + hh) * SEQ) + t) * HD + hd] = (bf16)o;
        }
      } else {  // MODE 2: v-projection -> (B,H,T,HD) bf16
        const int hh = n >> 7;
        const int hd = n & (HD - 1);
#pragma unroll
        for (int j = 0; j < 8; ++j) {
          const int m = mbase + j;
          const int b = m >> 10;
          const int t = m & (SEQ - 1);
          vws[(((size_t)(b * NH + hh) * SEQ) + t) * HD + hd] = (bf16)c[j];
        }
      }
    }
  }
}

// ======================================================================
// Flash-style causal attention. One WG per (b,h, 128-query block).
// 8 waves; each wave owns 16 query rows; d = 128.
// K tile staged via Tensor Data Mover (with LDS row padding); V tile
// transposed manually (TDM cannot transpose); P round-trips through LDS.
// ======================================================================
#define PVT 136   // padded 128-wide LDS row stride in halves (272 B)

__global__ __launch_bounds__(256)
void flash_attn_kernel(const bf16* __restrict__ qws, const bf16* __restrict__ kws,
                       const bf16* __restrict__ vws, float* __restrict__ yws)
{
  __shared__ bf16 Kt[HD * PVT];        // K tile  [key][d]   (TDM, padded rows)
  __shared__ bf16 Vt[HD * PVT];        // V^T tile [d][key]  (manual transpose)
  __shared__ bf16 Pb[8 * 16 * PVT];    // per-wave P [wave][qr][key]

  const int tid  = threadIdx.x;
  const int lane = tid & 31;
  const int wv   = tid >> 5;
  const int lr   = lane & 15;
  const int g8   = (lane >> 4) * 8;

  const int qb = blockIdx.x & 7;       // query block (128 rows)
  const int bh = blockIdx.x >> 3;      // 0..63
  const int b  = bh >> 4;
  const int h  = bh & 15;

  const bf16* qp = qws + (size_t)bh * SEQ * HD;
  const bf16* kp = kws + (size_t)bh * SEQ * HD;
  const bf16* vp = vws + (size_t)bh * SEQ * HD;

  const int qrow0 = qb * 128 + wv * 16;

  // Q fragments live in registers for the whole kernel (4 x K=32 steps)
  v16bf qf[4];
#pragma unroll
  for (int kk = 0; kk < 4; ++kk)
    qf[kk] = load_frag16(qp + (size_t)(qrow0 + lr) * HD + kk * 32 + g8);

  const v8f vzero = {0.f, 0.f, 0.f, 0.f, 0.f, 0.f, 0.f, 0.f};
  float mrow[8], lsum[8], alpha[8];
  v8f o[8];
#pragma unroll
  for (int j = 0; j < 8; ++j) { mrow[j] = -3.0e38f; lsum[j] = 0.f; }
#pragma unroll
  for (int dt = 0; dt < 8; ++dt) o[dt] = vzero;

  const float scale = 0.088388347648318447f;  // 1/sqrt(128)
  const u32 kt_lds = (u32)(uintptr_t)&Kt[0];  // LDS byte address (low 32 bits)

  for (int kbi = 0; kbi <= qb; ++kbi) {
    const int key0 = kbi * 128;

    // ---- K tile: one TDM copy per WG, overlapped with V staging below ----
    if (wv == 0) {
      // 128 rows x 128 bf16; pad 4 DWORDs (16 B) after every 64 DWORDs
      // (one 256 B row) -> padded LDS row stride 272 B == PVT halves.
      tdm_load_2d_bf16(kt_lds, kp + (size_t)key0 * HD,
                       /*rows=*/128, /*row_elems=*/128,
                       /*row_stride=*/HD,
                       /*pad_interval_code=*/5, /*pad_amount_code=*/3);
    }

    // ---- V^T staging: 4 keys x 8 d per task, vectorized b64 LDS stores ----
#pragma unroll
    for (int r = 0; r < 2; ++r) {
      const int task = tid + r * 256;       // 0..511
      const int kq = task >> 4;             // key quad 0..31
      const int dq = task & 15;             // d chunk of 8
      const bf16* src = vp + (size_t)(key0 + kq * 4) * HD + dq * 8;
      v8bf c0 = *(const v8bf*)(src);
      v8bf c1 = *(const v8bf*)(src + HD);
      v8bf c2 = *(const v8bf*)(src + 2 * HD);
      v8bf c3 = *(const v8bf*)(src + 3 * HD);
#pragma unroll
      for (int e = 0; e < 8; ++e) {
        v4bf tquad;
        tquad[0] = c0[e]; tquad[1] = c1[e]; tquad[2] = c2[e]; tquad[3] = c3[e];
        *(v4bf*)&Vt[(dq * 8 + e) * PVT + kq * 4] = tquad;
      }
    }

    if (wv == 0) __builtin_amdgcn_s_wait_tensorcnt(0);
    __syncthreads();                        // Kt + Vt ready for all waves

    // ---- S = Q K^T (16 x 128 per wave) from LDS K tile ----
    v8f s[8];
#pragma unroll
    for (int nt = 0; nt < 8; ++nt) {
      v8f a = vzero;
#pragma unroll
      for (int kk = 0; kk < 4; ++kk) {
        v16bf kf = load_frag16(&Kt[(nt * 16 + lr) * PVT + kk * 32 + g8]);
        a = wmma_bf16(qf[kk], kf, a);
      }
      s[nt] = a;
    }

    // ---- causal mask + scale + online softmax (16-lane reductions) ----
#pragma unroll
    for (int j = 0; j < 8; ++j) {
      const int qrow = qrow0 + g8 + j;
      float mx = -3.0e38f;
#pragma unroll
      for (int nt = 0; nt < 8; ++nt) {
        const int key = key0 + nt * 16 + lr;
        float sv = s[nt][j] * scale;
        sv = (key <= qrow) ? sv : -3.0e38f;
        s[nt][j] = sv;
        mx = fmaxf(mx, sv);
      }
      mx = fmaxf(mx, __shfl_xor(mx, 1, 32));
      mx = fmaxf(mx, __shfl_xor(mx, 2, 32));
      mx = fmaxf(mx, __shfl_xor(mx, 4, 32));
      mx = fmaxf(mx, __shfl_xor(mx, 8, 32));
      const float mnew = fmaxf(mrow[j], mx);
      alpha[j] = __expf(mrow[j] - mnew);
      mrow[j]  = mnew;
      float rs = 0.f;
#pragma unroll
      for (int nt = 0; nt < 8; ++nt) {
        const float p = __expf(s[nt][j] - mnew);
        s[nt][j] = p;
        rs += p;
      }
      rs += __shfl_xor(rs, 1, 32);
      rs += __shfl_xor(rs, 2, 32);
      rs += __shfl_xor(rs, 4, 32);
      rs += __shfl_xor(rs, 8, 32);
      lsum[j] = lsum[j] * alpha[j] + rs;
    }
#pragma unroll
    for (int dt = 0; dt < 8; ++dt)
#pragma unroll
      for (int j = 0; j < 8; ++j) o[dt][j] *= alpha[j];

    // ---- P -> per-wave LDS (C-layout -> A-layout conversion) ----
    bf16* pw = Pb + wv * 16 * PVT;
#pragma unroll
    for (int nt = 0; nt < 8; ++nt)
#pragma unroll
      for (int j = 0; j < 8; ++j)
        pw[(g8 + j) * PVT + nt * 16 + lr] = (bf16)s[nt][j];

    // ---- O += P @ V ----
#pragma unroll
    for (int kk = 0; kk < 4; ++kk) {
      v16bf pf = load_frag16(pw + (size_t)lr * PVT + kk * 32 + g8);
#pragma unroll
      for (int dt = 0; dt < 8; ++dt) {
        v16bf vf = load_frag16(&Vt[(dt * 16 + lr) * PVT + kk * 32 + g8]);
        o[dt] = wmma_bf16(pf, vf, o[dt]);
      }
    }
    __syncthreads();  // Kt/Vt reused next iteration
  }

  // ---- normalize + store y as f32 in (B, T, H*HD) layout ----
#pragma unroll
  for (int dt = 0; dt < 8; ++dt) {
    const int d = dt * 16 + lr;
#pragma unroll
    for (int j = 0; j < 8; ++j) {
      const int t = qrow0 + g8 + j;
      yws[((size_t)(b * SEQ + t) * DIM) + h * HD + d] = o[dt][j] / lsum[j];
    }
  }
}

// ======================================================================
// Host-side launch
// ======================================================================
extern "C" void kernel_launch(void* const* d_in, const int* in_sizes, int n_in,
                              void* d_out, int out_size, void* d_ws,
                              size_t ws_size, hipStream_t stream)
{
  (void)in_sizes; (void)n_in; (void)out_size; (void)ws_size;

  const float* v        = (const float*)d_in[0];
  const float* x_eps    = (const float*)d_in[1];
  const float* fcos     = (const float*)d_in[2];
  const float* fsin     = (const float*)d_in[3];
  const float* W_eps_at = (const float*)d_in[4];
  const float* W_attn   = (const float*)d_in[5];
  const float* W_proj   = (const float*)d_in[6];
  const float* W_eps_pr = (const float*)d_in[7];
  float* out = (float*)d_out;

  // Workspace layout (needs 80 MB):
  //   [0,16M)   q  bf16 (B,H,T,HD)
  //   [16,32M)  k  bf16 (B,H,T,HD)
  //   [32,48M)  v' bf16 (B,H,T,HD)
  //   [48,80M)  y  f32  (B,T,DIM)
  char* ws = (char*)d_ws;
  bf16*  qws = (bf16*)(ws);
  bf16*  kws = (bf16*)(ws + ((size_t)16 << 20));
  bf16*  vws = (bf16*)(ws + ((size_t)32 << 20));
  float* yws = (float*)(ws + ((size_t)48 << 20));

  // 1) qk = x_eps @ W_eps_attn (N=4096), fused RoPE + head transpose
  gemm_bf16_kernel<1><<<dim3((2 * DIM) / BN, MROWS / BM), 256, 0, stream>>>(
      x_eps, W_eps_at, 2 * DIM, DIM, nullptr, qws, kws, nullptr, fcos, fsin);

  // 2) vp = v @ W_attn, head transpose to bf16
  gemm_bf16_kernel<2><<<dim3(DIM / BN, MROWS / BM), 256, 0, stream>>>(
      v, W_attn, DIM, DIM, nullptr, nullptr, nullptr, vws, nullptr, nullptr);

  // 3) causal flash attention -> y (f32)
  flash_attn_kernel<<<dim3(BATCH * NH * (SEQ / 128)), 256, 0, stream>>>(
      qws, kws, vws, yws);

  // 4) v_out = y @ W_proj
  gemm_bf16_kernel<0><<<dim3(DIM / BN, MROWS / BM), 256, 0, stream>>>(
      yws, W_proj, DIM, DIM, out, nullptr, nullptr, nullptr, nullptr, nullptr);

  // 5) x_eps_out = x_eps @ W_eps_proj
  gemm_bf16_kernel<0><<<dim3(DIM / BN, MROWS / BM), 256, 0, stream>>>(
      x_eps, W_eps_pr, DIM, DIM, out + (size_t)MROWS * DIM,
      nullptr, nullptr, nullptr, nullptr, nullptr);
}
